// Network_23708219474702
// MI455X (gfx1250) — matrix operand order repaired and checked by reading, not compiled, not run on hardware
//
#include <hip/hip_runtime.h>
#include <hip/hip_bf16.h>

#define N_NODES  30000
#define N_EDGES  480000
#define N_GRAPHS 512
#define N_TILES  (N_EDGES / 16)
#define CONV_WAVES 4

typedef __attribute__((ext_vector_type(16))) __bf16 v16bf;
typedef __attribute__((ext_vector_type(8)))  float  v8f;

union ABf { unsigned int u[8]; v16bf v; };
static_assert(sizeof(ABf) == 32, "ABf must be 8 dwords");

// round-to-nearest-even f32 -> bf16, packed pair (lo, hi) into one dword
__device__ __forceinline__ unsigned int bf16pair(float a, float b) {
    unsigned int ua = __float_as_uint(a);
    ua += 0x7FFFu + ((ua >> 16) & 1u);
    unsigned int ub = __float_as_uint(b);
    ub += 0x7FFFu + ((ub >> 16) & 1u);
    return (ua >> 16) | (ub & 0xFFFF0000u);
}

// f32 -> bf16 broadcast into both halves of a dword
__device__ __forceinline__ unsigned int bf16bcast(float a) {
    unsigned int ua = __float_as_uint(a);
    ua += 0x7FFFu + ((ua >> 16) & 1u);
    unsigned int h = ua >> 16;
    return h | (h << 16);
}

// packed bf16 multiply (VOP3P V_PK_MUL_BF16) — one VALU op builds one
// A-fragment dword of the outer product.
__device__ __forceinline__ unsigned int pk_mul_bf16(unsigned int a, unsigned int b) {
    unsigned int d;
    asm("v_pk_mul_bf16 %0, %1, %2" : "=v"(d) : "v"(a), "v"(b));
    return d;
}

// ---------------------------------------------------------------------------
// Kernel: zero a float region
// ---------------------------------------------------------------------------
__global__ void zero_kernel(float* __restrict__ p, long long n) {
    long long i = (long long)blockIdx.x * blockDim.x + threadIdx.x;
    long long stride = (long long)gridDim.x * blockDim.x;
    for (; i < n; i += stride) p[i] = 0.0f;
}

// ---------------------------------------------------------------------------
// Kernel 1: per-edge geometry: sh (l<=3, 16 comps), radial embed,
//           atomic segment-sum of sh into x0[dst].
// ---------------------------------------------------------------------------
__global__ void edge_geom_kernel(const float* __restrict__ pos,
                                 const int* __restrict__ esrc,
                                 const int* __restrict__ edst,
                                 float* __restrict__ sh_out,
                                 float* __restrict__ scal_out,
                                 float* __restrict__ x0) {
    int e = blockIdx.x * blockDim.x + threadIdx.x;
    if (e >= N_EDGES) return;
    int s = esrc[e], d = edst[e];
    float ax = pos[3*s+0] - pos[3*d+0];
    float ay = pos[3*s+1] - pos[3*d+1];
    float az = pos[3*s+2] - pos[3*d+2];
    float r = sqrtf(ax*ax + ay*ay + az*az);
    float inv = 1.0f / fmaxf(r, 1e-9f);
    float x = ax*inv, y = ay*inv, z = az*inv;
    float x2 = x*x, y2 = y*y, z2 = z*z;
    float sh[16];
    sh[0]  = 1.0f;
    sh[1]  = 1.7320508f * x;
    sh[2]  = 1.7320508f * y;
    sh[3]  = 1.7320508f * z;
    sh[4]  = 3.8729833f * x * y;
    sh[5]  = 3.8729833f * y * z;
    sh[6]  = 1.1180340f * (3.0f*z2 - 1.0f);
    sh[7]  = 3.8729833f * x * z;
    sh[8]  = 1.9364917f * (x2 - y2);
    sh[9]  = 2.0916501f * y * (3.0f*x2 - y2);
    sh[10] = 10.2469508f * x * y * z;
    sh[11] = 1.6201852f * y * (5.0f*z2 - 1.0f);
    sh[12] = 1.3228757f * z * (5.0f*z2 - 3.0f);
    sh[13] = 1.6201852f * x * (5.0f*z2 - 1.0f);
    sh[14] = 5.1234754f * z * (x2 - y2);
    sh[15] = 2.0916501f * x * (x2 - y2);
#pragma unroll
    for (int j = 0; j < 16; j++) sh_out[(long long)e*16 + j] = sh[j];
    // radial embed * sqrt(3) folded into C
    const float C = 1.14136f * expf(2.0f) * 1.7320508f;
#pragma unroll
    for (int k = 0; k < 3; k++) {
        float uu = r - (0.5f + (float)k);
        float dd = uu*uu - 1.0f;
        scal_out[(long long)e*4 + k] = (dd < 0.0f) ? C * expf(1.0f/dd) : 0.0f;
    }
    scal_out[(long long)e*4 + 3] = 0.0f;
#pragma unroll
    for (int j = 0; j < 16; j++) atomicAdd(&x0[d*16 + j], sh[j]);
}

// ---------------------------------------------------------------------------
// Kernel 2: conv1 fused with weight-MLP (both GEMMs via bf16 WMMA).
//   ef(16e x 16n) = [outer(xs,sh) (16x256)] x [cg1^T half (256x80)]
//   w (16e x 16n) = [h=relu(scal@W1a)/s3 (16x256)] x [W2a half/16 (256x80)]
//   out += ef * w  -> atomic scatter to x1[dst]
//   'half' selects output columns [0,80) or [80,160).
// ---------------------------------------------------------------------------
__global__ __launch_bounds__(128) void conv1_kernel(
    const float* __restrict__ x0, const float* __restrict__ shbuf,
    const float* __restrict__ scal, const int* __restrict__ esrc,
    const int* __restrict__ edst, const float* __restrict__ W1a,
    const float* __restrict__ W2a, const float* __restrict__ cg1,
    float* __restrict__ x1, int half) {
    __shared__ unsigned int Bw[128][80];   // W2a pairs  (K/2=128, ncols=80)
    __shared__ unsigned int Bc[128][80];   // cg1 pairs
    __shared__ float        w1t[256][4];   // W1a transposed [col][row]
    __shared__ unsigned int xsb_t[CONV_WAVES][16][17]; // bf16 broadcast words
    __shared__ unsigned int shp_t[CONV_WAVES][16][9];  // bf16 sh pair words
    __shared__ float        sc_t[CONV_WAVES][16][4];
    __shared__ int          dst_t[CONV_WAVES][16];
    __shared__ unsigned int hp_t[CONV_WAVES][16][132]; // 128 used, padded

    const int tid = threadIdx.x;
    const float s_w2 = 1.0f / 16.0f;       // W2 / sqrt(256)
    const float s_cg = 0.51298918f;        // fold 1/sqrt(3.8) conv output scale
    for (int idx = tid; idx < 128 * 80; idx += blockDim.x) {
        int t = idx / 80, n = idx % 80;
        int ncol = half * 80 + n;
        Bw[t][n] = bf16pair(W2a[(2*t)*160 + ncol] * s_w2,
                            W2a[(2*t+1)*160 + ncol] * s_w2);
        Bc[t][n] = bf16pair(cg1[ncol*256 + 2*t] * s_cg,
                            cg1[ncol*256 + 2*t + 1] * s_cg);
    }
    for (int c = tid; c < 256; c += blockDim.x) {
        w1t[c][0] = W1a[c]; w1t[c][1] = W1a[256 + c]; w1t[c][2] = W1a[512 + c];
    }
    __syncthreads();

    const int wid = tid >> 5, lane = tid & 31;
    const int hi = lane >> 4, nl = lane & 15;
    const int gw = blockIdx.x * CONV_WAVES + wid;
    const int nW = gridDim.x * CONV_WAVES;
    const float inv_s38 = 0.51298918f;     // 1/sqrt(3.8) on gathered node feats
    const float inv_s3  = 0.57735027f;     // W1 / sqrt(3)

    for (int tile = gw; tile < N_TILES; tile += nW) {
        const int eb = tile * 16;
        if (lane < 16) {
            int e = eb + lane;
            dst_t[wid][lane] = edst[e];
            int s = esrc[e];
#pragma unroll
            for (int j = 0; j < 16; j++)
                xsb_t[wid][lane][j] = bf16bcast(x0[s*16 + j] * inv_s38);
#pragma unroll
            for (int q = 0; q < 8; q++)
                shp_t[wid][lane][q] = bf16pair(shbuf[(long long)e*16 + 2*q],
                                               shbuf[(long long)e*16 + 2*q + 1]);
            sc_t[wid][lane][0] = scal[(long long)e*4 + 0];
            sc_t[wid][lane][1] = scal[(long long)e*4 + 1];
            sc_t[wid][lane][2] = scal[(long long)e*4 + 2];
        }
        asm volatile("s_wait_dscnt 0" ::: "memory");
        // hidden layer h = relu(scal @ W1a) * inv_s3, bf16 pair-packed
        for (int idx = lane; idx < 16 * 128; idx += 32) {
            int m = idx >> 7, t = idx & 127;
            float s0 = sc_t[wid][m][0], s1 = sc_t[wid][m][1], s2 = sc_t[wid][m][2];
            float h0 = fmaxf(0.0f, s0*w1t[2*t][0]   + s1*w1t[2*t][1]   + s2*w1t[2*t][2])   * inv_s3;
            float h1 = fmaxf(0.0f, s0*w1t[2*t+1][0] + s1*w1t[2*t+1][1] + s2*w1t[2*t+1][2]) * inv_s3;
            hp_t[wid][m][t] = bf16pair(h0, h1);
        }
        asm volatile("s_wait_dscnt 0" ::: "memory");

        // hoist this lane's A-row operands into registers for the whole tile:
        // 16 xs broadcast words, 4 sh pair words ((v&3)+hi*4 — ks-independent)
        unsigned int xr[16];
#pragma unroll
        for (int q = 0; q < 16; q++) xr[q] = xsb_t[wid][nl][q];
        unsigned int shreg[4];
#pragma unroll
        for (int q = 0; q < 4; q++) shreg[q] = shp_t[wid][nl][q + hi*4];

#pragma unroll 1
        for (int nt = 0; nt < 5; nt++) {
            v8f accE = {0.f,0.f,0.f,0.f,0.f,0.f,0.f,0.f};
            v8f accW = {0.f,0.f,0.f,0.f,0.f,0.f,0.f,0.f};
#pragma unroll
            for (int ks = 0; ks < 8; ks++) {
                ABf aW, aO, bW, bC;
#pragma unroll
                for (int v = 0; v < 8; v++) {
                    int tb = ks*16 + ((v < 4) ? v : v + 4) + (hi ? 4 : 0); // B pair idx
                    bW.u[v] = Bw[tb][nt*16 + nl];
                    bC.u[v] = Bc[tb][nt*16 + nl];
                    int tg = ks*16 + ((v < 4) ? v : v + 4) + (hi ? 4 : 0); // A pair idx
                    aW.u[v] = hp_t[wid][nl][tg];
                    // outer-product fragment: i = 2*ks + (v>=4), shpair = (v&3)+hi*4
                    aO.u[v] = pk_mul_bf16(xr[2*ks + (v >> 2)], shreg[v & 3]);
                }
                accW = __builtin_amdgcn_wmma_f32_16x16x32_bf16(
                    false, aW.v, false, bW.v, (short)0, accW, false, false);
                accE = __builtin_amdgcn_wmma_f32_16x16x32_bf16(
                    false, aO.v, false, bC.v, (short)0, accE, false, false);
            }
#pragma unroll
            for (int r = 0; r < 8; r++) {
                int m = r + (hi ? 8 : 0);
                atomicAdd(&x1[dst_t[wid][m]*160 + half*80 + nt*16 + nl],
                          accE[r] * accW[r]);
            }
        }
    }
}

// ---------------------------------------------------------------------------
// Kernel 3: gate  x1(N,160) -> xg(N,128)
// ---------------------------------------------------------------------------
__global__ void gate_kernel(const float* __restrict__ x1, float* __restrict__ xg) {
    int idx = blockIdx.x * blockDim.x + threadIdx.x;
    if (idx >= N_NODES * 128) return;
    int n = idx >> 7, j = idx & 127;
    const float* xr = x1 + n * 160;
    float v;
    if (j < 16)       v = fmaxf(0.0f, xr[j]);
    else if (j < 32)  v = fabsf(xr[j]);
    else {
        int jj = j - 32;
        int c = jj / 3;
        float gs = xr[32 + c];
        float g = ((c >> 3) & 1) ? tanhf(gs) : fmaxf(0.0f, gs);
        v = xr[64 + jj] * g;
    }
    xg[idx] = v;
}

// ---------------------------------------------------------------------------
// Kernel 4: conv2 fused with second weight-MLP (bf16 WMMA, N=7 padded to 16).
//   ef(16e x 7) = [outer(xg,sh) (16x2048)] x [cg2^T (2048x7)]
//   w (16e x 7) = [h2 (16x256)] x [W2b/16 (256x7)]
// ---------------------------------------------------------------------------
__global__ __launch_bounds__(128) void conv2_kernel(
    const float* __restrict__ xg, const float* __restrict__ shbuf,
    const float* __restrict__ scal, const int* __restrict__ esrc,
    const int* __restrict__ edst, const float* __restrict__ W1b,
    const float* __restrict__ W2b, const float* __restrict__ cg2,
    float* __restrict__ x2) {
    __shared__ unsigned int Bc[1024][8];   // cg2 pairs, col 7 zeroed
    __shared__ unsigned int Bw[128][8];    // W2b pairs, col 7 zeroed
    __shared__ float        w1t[256][4];
    __shared__ unsigned int xsb_t[CONV_WAVES][16][132]; // bf16 broadcast words
    __shared__ unsigned int shp_t[CONV_WAVES][16][9];   // bf16 sh pair words
    __shared__ float        sc_t[CONV_WAVES][16][4];
    __shared__ int          src_t[CONV_WAVES][16];
    __shared__ int          dst_t[CONV_WAVES][16];
    __shared__ unsigned int hp_t[CONV_WAVES][16][132];

    const int tid = threadIdx.x;
    const float s_w2 = 1.0f / 16.0f;
    const float s_cg = 0.51298918f;
    for (int idx = tid; idx < 1024 * 8; idx += blockDim.x) {
        int t = idx >> 3, n = idx & 7;
        Bc[t][n] = (n < 7) ? bf16pair(cg2[n*2048 + 2*t] * s_cg,
                                      cg2[n*2048 + 2*t + 1] * s_cg) : 0u;
    }
    for (int idx = tid; idx < 128 * 8; idx += blockDim.x) {
        int t = idx >> 3, n = idx & 7;
        Bw[t][n] = (n < 7) ? bf16pair(W2b[(2*t)*7 + n] * s_w2,
                                      W2b[(2*t+1)*7 + n] * s_w2) : 0u;
    }
    for (int c = tid; c < 256; c += blockDim.x) {
        w1t[c][0] = W1b[c]; w1t[c][1] = W1b[256 + c]; w1t[c][2] = W1b[512 + c];
    }
    __syncthreads();

    const int wid = tid >> 5, lane = tid & 31;
    const int hi = lane >> 4, nl = lane & 15;
    const int ncol = (nl < 7) ? nl : 7;  // lanes 7..15 read zeroed column
    const int gw = blockIdx.x * CONV_WAVES + wid;
    const int nW = gridDim.x * CONV_WAVES;
    const float inv_s3 = 0.57735027f;

    for (int tile = gw; tile < N_TILES; tile += nW) {
        const int eb = tile * 16;
        if (lane < 16) {
            int e = eb + lane;
            src_t[wid][lane] = esrc[e];
            dst_t[wid][lane] = edst[e];
#pragma unroll
            for (int q = 0; q < 8; q++)
                shp_t[wid][lane][q] = bf16pair(shbuf[(long long)e*16 + 2*q],
                                               shbuf[(long long)e*16 + 2*q + 1]);
            sc_t[wid][lane][0] = scal[(long long)e*4 + 0];
            sc_t[wid][lane][1] = scal[(long long)e*4 + 1];
            sc_t[wid][lane][2] = scal[(long long)e*4 + 2];
        }
        asm volatile("s_wait_dscnt 0" ::: "memory");
        for (int idx = lane; idx < 16 * 128; idx += 32) {
            int m = idx >> 7, t = idx & 127;
            xsb_t[wid][m][t] = bf16bcast(xg[src_t[wid][m]*128 + t]);
            float s0 = sc_t[wid][m][0], s1 = sc_t[wid][m][1], s2 = sc_t[wid][m][2];
            float h0 = fmaxf(0.0f, s0*w1t[2*t][0]   + s1*w1t[2*t][1]   + s2*w1t[2*t][2])   * inv_s3;
            float h1 = fmaxf(0.0f, s0*w1t[2*t+1][0] + s1*w1t[2*t+1][1] + s2*w1t[2*t+1][2]) * inv_s3;
            hp_t[wid][m][t] = bf16pair(h0, h1);
        }
        asm volatile("s_wait_dscnt 0" ::: "memory");

        unsigned int shreg[4];
#pragma unroll
        for (int q = 0; q < 4; q++) shreg[q] = shp_t[wid][nl][q + hi*4];

        v8f accW = {0.f,0.f,0.f,0.f,0.f,0.f,0.f,0.f};
#pragma unroll
        for (int ks = 0; ks < 8; ks++) {
            ABf aW, bW;
#pragma unroll
            for (int v = 0; v < 8; v++) {
                bW.u[v] = Bw[ks*16 + ((v < 4) ? v : v + 4) + (hi ? 4 : 0)][ncol];
                int tg = ks*16 + ((v < 4) ? v : v + 4) + (hi ? 4 : 0);
                aW.u[v] = hp_t[wid][nl][tg];
            }
            accW = __builtin_amdgcn_wmma_f32_16x16x32_bf16(
                false, aW.v, false, bW.v, (short)0, accW, false, false);
        }

        v8f accE = {0.f,0.f,0.f,0.f,0.f,0.f,0.f,0.f};
#pragma unroll 4
        for (int ks = 0; ks < 64; ks++) {
            ABf aO, bC;
            unsigned int xb0 = xsb_t[wid][nl][2*ks];
            unsigned int xb1 = xsb_t[wid][nl][2*ks + 1];
#pragma unroll
            for (int v = 0; v < 8; v++) {
                bC.u[v] = Bc[ks*16 + ((v < 4) ? v : v + 4) + (hi ? 4 : 0)][ncol];
                aO.u[v] = pk_mul_bf16((v < 4) ? xb0 : xb1, shreg[v & 3]);
            }
            accE = __builtin_amdgcn_wmma_f32_16x16x32_bf16(
                false, aO.v, false, bC.v, (short)0, accE, false, false);
        }
        if (nl < 7) {
#pragma unroll
            for (int r = 0; r < 8; r++) {
                int m = r + (hi ? 8 : 0);
                atomicAdd(&x2[dst_t[wid][m]*8 + nl], accE[r] * accW[r]);
            }
        }
    }
}

// ---------------------------------------------------------------------------
// Kernel 5: per-graph segment sum, out = sum(x2)/2
// ---------------------------------------------------------------------------
__global__ void reduce_kernel(const float* __restrict__ x2,
                              const int* __restrict__ batch,
                              float* __restrict__ out) {
    int idx = blockIdx.x * blockDim.x + threadIdx.x;
    if (idx >= N_NODES * 8) return;
    int n = idx >> 3, k = idx & 7;
    if (k < 7) atomicAdd(&out[batch[n]*7 + k], x2[idx] * 0.5f);
}

// ---------------------------------------------------------------------------
extern "C" void kernel_launch(void* const* d_in, const int* in_sizes, int n_in,
                              void* d_out, int out_size, void* d_ws, size_t ws_size,
                              hipStream_t stream) {
    (void)in_sizes; (void)n_in; (void)out_size; (void)ws_size;
    const float* pos  = (const float*)d_in[0];
    const float* W1a  = (const float*)d_in[1];
    const float* W2a  = (const float*)d_in[2];
    const float* cg1  = (const float*)d_in[3];
    const float* W1b  = (const float*)d_in[4];
    const float* W2b  = (const float*)d_in[5];
    const float* cg2  = (const float*)d_in[6];
    const int* esrc   = (const int*)d_in[7];
    const int* edst   = (const int*)d_in[8];
    const int* batch  = (const int*)d_in[9];
    float* out = (float*)d_out;
    float* ws  = (float*)d_ws;

    // workspace layout (floats)
    float* x0   = ws;               // N*16        =   480,000
    float* x1   = ws + 480000;      // N*160       = 4,800,000
    float* x2   = ws + 5280000;     // N*8         =   240,000
    float* shb  = ws + 5520000;     // E*16        = 7,680,000
    float* scal = ws + 13200000;    // E*4         = 1,920,000
    float* xg   = ws + 15120000;    // N*128       = 3,840,000

    zero_kernel<<<2048, 256, 0, stream>>>(ws, 5520000LL);              // x0,x1,x2
    zero_kernel<<<16, 256, 0, stream>>>(out, (long long)(N_GRAPHS * 7));

    edge_geom_kernel<<<(N_EDGES + 255) / 256, 256, 0, stream>>>(
        pos, esrc, edst, shb, scal, x0);

    conv1_kernel<<<1024, 128, 0, stream>>>(x0, shb, scal, esrc, edst,
                                           W1a, W2a, cg1, x1, 0);
    conv1_kernel<<<1024, 128, 0, stream>>>(x0, shb, scal, esrc, edst,
                                           W1a, W2a, cg1, x1, 1);

    gate_kernel<<<(N_NODES * 128 + 255) / 256, 256, 0, stream>>>(x1, xg);

    conv2_kernel<<<1024, 128, 0, stream>>>(xg, shb, scal, esrc, edst,
                                           W1b, W2b, cg2, x2);

    reduce_kernel<<<(N_NODES * 8 + 255) / 256, 256, 0, stream>>>(x2, batch, out);
}
